// LinearAttention_25812753449190
// MI455X (gfx1250) — compile-verified, matrix operand order checked by reading, not run on hardware
//
#include <hip/hip_runtime.h>

typedef __attribute__((ext_vector_type(16))) _Float16 v16h;
typedef __attribute__((ext_vector_type(2)))  __fp16   v2fp;   // return type of cvt_pkrtz
typedef __attribute__((ext_vector_type(8)))  float    v8f;
typedef unsigned int   u32;
typedef unsigned short u16;

#define LDSPAD 4

__device__ __forceinline__ u16 f16bits(float f) {
  union { _Float16 h; u16 u; } cv; cv.h = (_Float16)f; return cv.u;
}
__device__ __forceinline__ float h2f(u16 b) {
  union { _Float16 h; u16 u; } cv; cv.u = b; return (float)cv.h;
}
// pack two f32 -> one dword of two f16 (v_cvt_pk_rtz_f16_f32)
__device__ __forceinline__ u32 pkh2(float a, float b) {
  union { v2fp h; u32 u; } cv; cv.h = __builtin_amdgcn_cvt_pkrtz(a, b); return cv.u;
}

// Gather one 16x32 f16 A/B fragment from an LDS tile stored [row][k] (ld in u16
// units, even; kofs even). Per CDNA5 ISA 16-bit A layout:
//   lane l holds row (l&15); element e holds K = (e/8)*16 + (l/16)*8 + (e%8).
// B (32x16) mirrors this with "row" = output column (tile staged transposed).
__device__ __forceinline__ v16h ldsFrag(const u16* lds, int row, int ld, int kofs, int lane) {
  union { u32 u[8]; v16h v; } f;
  const int base = row * ld + kofs + ((lane >> 4) << 3);
#pragma unroll
  for (int p = 0; p < 8; ++p) {
    const int kk = ((p >> 2) << 4) + ((p & 3) << 1);
    f.u[p] = *(const u32*)(lds + base + kk);
  }
  return f.v;
}

__device__ __forceinline__ v8f wmma_f16(v16h a, v16h b, v8f c) {
  return __builtin_amdgcn_wmma_f32_16x16x32_f16(false, a, false, b, (short)0, c, false, false);
}

// ---------------------------------------------------------------------------
// GEMM: C[M,512] = fmap(A[M,512] @ W[512,512] + bias). BM=BN=128, BK=32,
// 256 threads = 8 wave32s in a 4(m) x 2(n) grid; each wave: 2x4 WMMA tiles.
// ---------------------------------------------------------------------------
template<int FMAP, bool IN_HALF, bool OUT_FLOAT>
__global__ __launch_bounds__(256)
void gemm_proj(const void* __restrict__ Ain, const float* __restrict__ W,
               const float* __restrict__ bias, void* __restrict__ Cout) {
  constexpr int BM = 128, BN = 128, BK = 32, LDT = BK + LDSPAD;
  __shared__ u16 At[BM * LDT];
  __shared__ u16 Bt[BN * LDT];

  const int t = threadIdx.x;
  const int lane = t & 31, wid = t >> 5;
  const int waveM = wid >> 1, waveN = wid & 1;
  const int m0 = blockIdx.y * BM, n0 = blockIdx.x * BN;

  const v8f vz = {0.f,0.f,0.f,0.f,0.f,0.f,0.f,0.f};
  v8f c[2][4];
#pragma unroll
  for (int i = 0; i < 2; ++i)
#pragma unroll
    for (int j = 0; j < 4; ++j) c[i][j] = vz;

  const int arow = t >> 1, acol = (t & 1) * 16;     // A: 128 rows x 32 k
  const int bkp  = (t & 15) * 2, bng = (t >> 4) * 8; // B: (k,k+1) pair x 8 n cols

  for (int k0 = 0; k0 < 512; k0 += BK) {
    // ---- stage A tile (packed dword stores) ----
    if (IN_HALF) {
      const _Float16* Ah = (const _Float16*)Ain;
      const u32* src = (const u32*)(Ah + (size_t)(m0 + arow) * 512 + k0 + acol);
#pragma unroll
      for (int i = 0; i < 8; ++i)
        *(u32*)&At[arow * LDT + acol + i * 2] = src[i];
      if (k0 + BK < 512)
        __builtin_prefetch((const void*)(src + 8), 0, 1);
    } else {
      const float* Af = (const float*)Ain;
      const float4* src = (const float4*)(Af + (size_t)(m0 + arow) * 512 + k0 + acol);
#pragma unroll
      for (int i = 0; i < 4; ++i) {
        const float4 q = src[i];
        *(u32*)&At[arow * LDT + acol + i*4 + 0] = pkh2(q.x, q.y);
        *(u32*)&At[arow * LDT + acol + i*4 + 2] = pkh2(q.z, q.w);
      }
      if (k0 + BK < 512)
        __builtin_prefetch((const void*)(src + 8), 0, 1);
    }
    // ---- stage B tile transposed: pack along k so LDS stores are dwords ----
    {
      const float4* r0 = (const float4*)(W + (size_t)(k0 + bkp) * 512 + n0 + bng);
      const float4* r1 = (const float4*)(W + (size_t)(k0 + bkp + 1) * 512 + n0 + bng);
      const float4 a0 = r0[0], a1 = r0[1];
      const float4 b0 = r1[0], b1 = r1[1];
      *(u32*)&Bt[(bng + 0) * LDT + bkp] = pkh2(a0.x, b0.x);
      *(u32*)&Bt[(bng + 1) * LDT + bkp] = pkh2(a0.y, b0.y);
      *(u32*)&Bt[(bng + 2) * LDT + bkp] = pkh2(a0.z, b0.z);
      *(u32*)&Bt[(bng + 3) * LDT + bkp] = pkh2(a0.w, b0.w);
      *(u32*)&Bt[(bng + 4) * LDT + bkp] = pkh2(a1.x, b1.x);
      *(u32*)&Bt[(bng + 5) * LDT + bkp] = pkh2(a1.y, b1.y);
      *(u32*)&Bt[(bng + 6) * LDT + bkp] = pkh2(a1.z, b1.z);
      *(u32*)&Bt[(bng + 7) * LDT + bkp] = pkh2(a1.w, b1.w);
      if (k0 + BK < 512)
        __builtin_prefetch((const void*)(W + (size_t)(k0 + BK + bkp) * 512 + n0 + bng), 0, 1);
    }
    __syncthreads();

    v16h a[2], b[4];
#pragma unroll
    for (int mi = 0; mi < 2; ++mi)
      a[mi] = ldsFrag(At, waveM * 32 + mi * 16 + (lane & 15), LDT, 0, lane);
#pragma unroll
    for (int ni = 0; ni < 4; ++ni)
      b[ni] = ldsFrag(Bt, waveN * 64 + ni * 16 + (lane & 15), LDT, 0, lane);
#pragma unroll
    for (int mi = 0; mi < 2; ++mi)
#pragma unroll
      for (int ni = 0; ni < 4; ++ni)
        c[mi][ni] = wmma_f16(a[mi], b[ni], c[mi][ni]);
    __syncthreads();
  }

#pragma unroll
  for (int mi = 0; mi < 2; ++mi)
#pragma unroll
    for (int ni = 0; ni < 4; ++ni) {
      const int n = n0 + waveN * 64 + ni * 16 + (lane & 15);
      const float bb = bias[n];
#pragma unroll
      for (int r = 0; r < 8; ++r) {
        const int m = m0 + waveM * 32 + mi * 16 + ((lane >> 4) << 3) + r;
        float v = c[mi][ni][r] + bb;
        if (FMAP) v = (v > 0.f) ? (v + 1.f) : __expf(v);   // elu(x)+1
        if (OUT_FLOAT) ((float*)Cout)[(size_t)m * 512 + n] = v;
        else ((_Float16*)Cout)[(size_t)m * 512 + n] = (_Float16)v;
      }
    }
}

// ---------------------------------------------------------------------------
// ksum[b,h,d] = sum_s K[b,s,h,d]. Vectorized uint4 loads (8 halves of d per
// load), 32-way s-split, ds_add_f32 reduction.
// ---------------------------------------------------------------------------
__global__ __launch_bounds__(256)
void ksum_kernel(const _Float16* __restrict__ K, float* __restrict__ ksum) {
  const int bh = blockIdx.x, b = bh >> 3, h = bh & 7;
  const int t = threadIdx.x;
  const int d8 = (t & 7) * 8, sp = t >> 3;   // 32-way split over s
  __shared__ float red[64];
  if (t < 64) red[t] = 0.f;
  __syncthreads();
  float acc[8] = {0.f,0.f,0.f,0.f,0.f,0.f,0.f,0.f};
  const _Float16* base = K + (size_t)b * 4096 * 512 + h * 64 + d8;
  for (int s = sp; s < 4096; s += 32) {
    const uint4 q = *(const uint4*)(base + (size_t)s * 512);
    const u16* p = (const u16*)&q;
#pragma unroll
    for (int i = 0; i < 8; ++i) acc[i] += h2f(p[i]);
  }
#pragma unroll
  for (int i = 0; i < 8; ++i) atomicAdd(&red[d8 + i], acc[i]);
  __syncthreads();
  if (t < 64) ksum[bh * 64 + t] = red[t];
}

// ---------------------------------------------------------------------------
// kv[b,h,m,d] = sum_s V[b,s,h,m] * K[b,s,h,d]  -- per (b,h): Vt[64xS] @ K[Sx64]
// 8 waves split S; each wave owns the full 4x4 WMMA tile grid; LDS atomic
// reduction (ds_add_f32) across waves. Staging packs (s,s+1) f16 pairs so the
// transposed scatter uses dword LDS stores.
// ---------------------------------------------------------------------------
__global__ __launch_bounds__(256)
void kv_kernel(const _Float16* __restrict__ Kf, const _Float16* __restrict__ Vf,
               float* __restrict__ kvg) {
  constexpr int SK = 256, LDT = SK + LDSPAD;
  __shared__ u16 Vt[64 * LDT];
  __shared__ u16 Kt[64 * LDT];
  __shared__ float cacc[64 * 64];

  const int bh = blockIdx.x, b = bh >> 3, h = bh & 7;
  const int t = threadIdx.x, lane = t & 31, wid = t >> 5;

  const v8f vz = {0.f,0.f,0.f,0.f,0.f,0.f,0.f,0.f};
  v8f c[4][4];
#pragma unroll
  for (int i = 0; i < 4; ++i)
#pragma unroll
    for (int j = 0; j < 4; ++j) c[i][j] = vz;

  const int m8 = (t & 7) * 8, sp2 = (t >> 3) * 2;  // (s,s+1) pair per thread

  for (int it = 0; it < 16; ++it) {
    const int sbase = b * 4096 + it * 256;
#pragma unroll
    for (int p = 0; p < 4; ++p) {
      const int sl = sp2 + p * 64;
      const size_t g0 = (size_t)(sbase + sl) * 512 + h * 64 + m8;
      const uint4 v0 = *(const uint4*)(Vf + g0);
      const uint4 v1 = *(const uint4*)(Vf + g0 + 512);
      const uint4 k0 = *(const uint4*)(Kf + g0);
      const uint4 k1 = *(const uint4*)(Kf + g0 + 512);
      const u16* vp0 = (const u16*)&v0; const u16* vp1 = (const u16*)&v1;
      const u16* kp0 = (const u16*)&k0; const u16* kp1 = (const u16*)&k1;
#pragma unroll
      for (int i = 0; i < 8; ++i) {
        *(u32*)&Vt[(m8 + i) * LDT + sl] = (u32)vp0[i] | ((u32)vp1[i] << 16);
        *(u32*)&Kt[(m8 + i) * LDT + sl] = (u32)kp0[i] | ((u32)kp1[i] << 16);
      }
    }
    __syncthreads();

    const int kofs = wid * 32;   // per-wave K-split within the staged 256
    v16h a[4], bf[4];
#pragma unroll
    for (int mi = 0; mi < 4; ++mi)
      a[mi] = ldsFrag(Vt, mi * 16 + (lane & 15), LDT, kofs, lane);
#pragma unroll
    for (int ni = 0; ni < 4; ++ni)
      bf[ni] = ldsFrag(Kt, ni * 16 + (lane & 15), LDT, kofs, lane);
#pragma unroll
    for (int mi = 0; mi < 4; ++mi)
#pragma unroll
      for (int ni = 0; ni < 4; ++ni)
        c[mi][ni] = wmma_f16(a[mi], bf[ni], c[mi][ni]);
    __syncthreads();
  }

  for (int i = t; i < 4096; i += 256) cacc[i] = 0.f;
  __syncthreads();
#pragma unroll
  for (int mi = 0; mi < 4; ++mi)
#pragma unroll
    for (int ni = 0; ni < 4; ++ni) {
      const int n = ni * 16 + (lane & 15);
#pragma unroll
      for (int r = 0; r < 8; ++r) {
        const int m = mi * 16 + ((lane >> 4) << 3) + r;
        atomicAdd(&cacc[m * 64 + n], c[mi][ni][r]);
      }
    }
  __syncthreads();
  for (int i = t; i < 4096; i += 256) kvg[(size_t)bh * 4096 + i] = cacc[i];
}

// ---------------------------------------------------------------------------
// out[b,s,h,m] = z * sum_d Q[b,s,h,d] * kv[b,h,m,d],  z = 1/(Q.ksum + eps)
// Block: 128 rows of s for one (b,h). 8 waves in 4(m) x 2(n); N=64, K=64.
// ---------------------------------------------------------------------------
__global__ __launch_bounds__(256)
void attn_kernel(const _Float16* __restrict__ Qf, const float* __restrict__ kvg,
                 const float* __restrict__ ksumg, _Float16* __restrict__ Af) {
  constexpr int LDQ = 64 + LDSPAD;
  __shared__ u16 Qt[128 * LDQ];
  __shared__ u16 KVt[64 * LDQ];
  __shared__ float ksumLds[64];
  __shared__ float zLds[128];

  const int st = blockIdx.x, h = blockIdx.y, b = blockIdx.z;
  const int bh = b * 8 + h;
  const int t = threadIdx.x, lane = t & 31, wid = t >> 5;
  const int waveM = wid >> 1, waveN = wid & 1;
  const int s0 = st * 128;

  // stage Q [128][64] (dword copies, already f16)
  const int qrow = t >> 1, qcol = (t & 1) * 32;
  const u32* qsrc = (const u32*)(Qf + (size_t)(b * 4096 + s0 + qrow) * 512 + h * 64 + qcol);
#pragma unroll
  for (int i = 0; i < 16; ++i)
    *(u32*)&Qt[qrow * LDQ + qcol + i * 2] = qsrc[i];
  // stage kv [m][d] -> B-style [n=m][k=d]; pack f32 pairs -> dword stores
  {
    const int flat0 = t * 16;             // 16 contiguous d within one m-row
    const int mrow = flat0 >> 6, dcol = flat0 & 63;
    const float* src = kvg + (size_t)bh * 4096 + flat0;
#pragma unroll
    for (int i = 0; i < 8; ++i)
      *(u32*)&KVt[mrow * LDQ + dcol + i * 2] = pkh2(src[i * 2], src[i * 2 + 1]);
  }
  if (t < 64) ksumLds[t] = ksumg[bh * 64 + t];
  __syncthreads();

  if (t < 128) {
    float dot = 0.f;
#pragma unroll
    for (int d = 0; d < 64; ++d) dot += h2f(Qt[t * LDQ + d]) * ksumLds[d];
    zLds[t] = 1.f / (dot + 1e-6f);
  }

  const v8f vz = {0.f,0.f,0.f,0.f,0.f,0.f,0.f,0.f};
  v8f c[2][2];
#pragma unroll
  for (int i = 0; i < 2; ++i)
#pragma unroll
    for (int j = 0; j < 2; ++j) c[i][j] = vz;

#pragma unroll
  for (int k0 = 0; k0 < 64; k0 += 32) {
    v16h a[2], bf[2];
#pragma unroll
    for (int mi = 0; mi < 2; ++mi)
      a[mi] = ldsFrag(Qt, waveM * 32 + mi * 16 + (lane & 15), LDQ, k0, lane);
#pragma unroll
    for (int ni = 0; ni < 2; ++ni)
      bf[ni] = ldsFrag(KVt, waveN * 32 + ni * 16 + (lane & 15), LDQ, k0, lane);
#pragma unroll
    for (int mi = 0; mi < 2; ++mi)
#pragma unroll
      for (int ni = 0; ni < 2; ++ni)
        c[mi][ni] = wmma_f16(a[mi], bf[ni], c[mi][ni]);
  }
  __syncthreads();   // zLds ready

#pragma unroll
  for (int mi = 0; mi < 2; ++mi)
#pragma unroll
    for (int ni = 0; ni < 2; ++ni) {
      const int n = waveN * 32 + ni * 16 + (lane & 15);
#pragma unroll
      for (int r = 0; r < 8; ++r) {
        const int m = waveM * 32 + mi * 16 + ((lane >> 4) << 3) + r;
        const float v = c[mi][ni][r] * zLds[m];
        Af[(size_t)(b * 4096 + s0 + m) * 512 + h * 64 + n] = (_Float16)v;
      }
    }
}

// ---------------------------------------------------------------------------
extern "C" void kernel_launch(void* const* d_in, const int* in_sizes, int n_in,
                              void* d_out, int out_size, void* d_ws, size_t ws_size,
                              hipStream_t stream) {
  const float* x  = (const float*)d_in[0];
  const float* Wq = (const float*)d_in[1];
  const float* bq = (const float*)d_in[2];
  const float* Wk = (const float*)d_in[3];
  const float* bk = (const float*)d_in[4];
  const float* Wv = (const float*)d_in[5];
  const float* bv = (const float*)d_in[6];
  const float* Wo = (const float*)d_in[7];
  const float* bo = (const float*)d_in[8];

  const size_t MT = (size_t)8 * 4096;           // 32768 token rows
  _Float16* Qws = (_Float16*)d_ws;              // 32 MB each, L2-resident
  _Float16* Kws = Qws + MT * 512;
  _Float16* Vws = Kws + MT * 512;
  _Float16* Aws = Vws + MT * 512;
  float* kvg    = (float*)(Aws + MT * 512);     // 64 x 64 x 64 = 1 MB
  float* ksumg  = kvg + (size_t)64 * 4096;      // 16 KB

  dim3 blk(256);
  dim3 gP(4, 256);                              // 512/128 n-tiles x 32768/128 m-tiles

  gemm_proj<1, false, false><<<gP, blk, 0, stream>>>(x, Wq, bq, Qws); // q = fmap(xWq+bq)
  gemm_proj<1, false, false><<<gP, blk, 0, stream>>>(x, Wk, bk, Kws); // k = fmap(xWk+bk)
  gemm_proj<0, false, false><<<gP, blk, 0, stream>>>(x, Wv, bv, Vws); // v = xWv+bv

  ksum_kernel<<<64, blk, 0, stream>>>(Kws, ksumg);
  kv_kernel<<<64, blk, 0, stream>>>(Kws, Vws, kvg);

  attn_kernel<<<dim3(32, 8, 8), blk, 0, stream>>>(Qws, kvg, ksumg, Aws);

  gemm_proj<0, true, true><<<gP, blk, 0, stream>>>(Aws, Wo, bo, (float*)d_out);
}
